// MultiHeadAttention_23450521436219
// MI455X (gfx1250) — matrix operand order repaired and checked by reading, not compiled, not run on hardware
//
#include <hip/hip_runtime.h>
#include <hip/hip_bf16.h>

#define BATCH  2
#define SEQ    2048
#define DMODEL 1024
#define NHEAD  16
#define DHEAD  64

typedef __attribute__((ext_vector_type(16))) __bf16 v16bf;
typedef __attribute__((ext_vector_type(8)))  __bf16 v8bf;
typedef __attribute__((ext_vector_type(8)))  float  v8f;

__device__ __forceinline__ v8f wmma_bf16(v16bf a, v16bf b, v8f c) {
  // D = A(16x32 bf16) x B(32x16 bf16) + C(16x16 f32)
  return __builtin_amdgcn_wmma_f32_16x16x32_bf16(false, a, false, b,
                                                 (short)0, c, false, false);
}

// ---------------------------------------------------------------------------
// f32 -> bf16 pack (weights): one-time conversion so GEMM B-fragments are
// plain 32-byte loads with zero VALU conversion in the hot loop.
// ---------------------------------------------------------------------------
__global__ __launch_bounds__(256)
void pack_bf16_kernel(const float* __restrict__ x, __bf16* __restrict__ y, int n) {
  const int i = (blockIdx.x * 256 + threadIdx.x) * 8;
  if (i < n) {
    v8f v = *(const v8f*)(x + i);
    v8bf o;
#pragma unroll
    for (int j = 0; j < 8; ++j) o[j] = (__bf16)v[j];
    *(v8bf*)(y + i) = o;
  }
}

// ---------------------------------------------------------------------------
// GEMM: Y[M,N] = X[M,K] @ W[N,K]^T  (M=4096, N=K=1024), W pre-packed bf16.
// Wave owns a 16x64 tile; all 4 B-fragments are loaded as one clause before
// the 4 back-to-back WMMAs (no per-WMMA load stall).
// ---------------------------------------------------------------------------
template <bool A_BF16, bool OUT_F32>
__global__ __launch_bounds__(256)
void proj_gemm_kernel(const void* __restrict__ xv, const __bf16* __restrict__ w,
                      void* __restrict__ yv, int store_mode) {
  const int lane  = threadIdx.x & 31;
  const int wave  = threadIdx.x >> 5;
  const int task  = blockIdx.x * 8 + wave;   // 4096 wave-tasks
  const int mbase = (task >> 4) * 16;
  const int nbase = (task & 15) * 64;
  const int m     = lane & 15;
  const int hi    = lane >> 4;

  const float*  xf = (const float*)xv;
  const __bf16* xb = (const __bf16*)xv;

  v8f acc[4] = {};

  for (int kc = 0; kc < DMODEL; kc += 32) {
    // A fragment (16x32): lane m holds K = hi*16..+15
    v16bf afrag;
    if constexpr (A_BF16) {
      const __bf16* p = xb + (size_t)(mbase + m) * DMODEL + kc;
      v8bf lo = *(const v8bf*)(p + hi * 16);
      v8bf hh = *(const v8bf*)(p + hi * 16 + 8);
#pragma unroll
      for (int i = 0; i < 8; ++i) { afrag[i] = lo[i]; afrag[8 + i] = hh[i]; }
    } else {
      const float* p = xf + (size_t)(mbase + m) * DMODEL + kc;
      v8f lo = *(const v8f*)(p + hi * 16);
      v8f hh = *(const v8f*)(p + hi * 16 + 8);
#pragma unroll
      for (int i = 0; i < 8; ++i) {
        afrag[i] = (__bf16)lo[i]; afrag[8 + i] = (__bf16)hh[i];
      }
    }
    // 4 B fragments in one load clause, then 4 WMMAs back-to-back
    v16bf bfrag[4];
#pragma unroll
    for (int nt = 0; nt < 4; ++nt)
      bfrag[nt] = *(const v16bf*)(w + (size_t)(nbase + nt * 16 + m) * DMODEL +
                                  kc + hi * 16);
#pragma unroll
    for (int nt = 0; nt < 4; ++nt)
      acc[nt] = wmma_bf16(afrag, bfrag[nt], acc[nt]);
  }

  // C element j -> (M = j+8*hi, N = lane&15)
#pragma unroll
  for (int nt = 0; nt < 4; ++nt) {
#pragma unroll
    for (int j = 0; j < 8; ++j) {
      const int row = mbase + j + 8 * hi;
      const int col = nbase + nt * 16 + m;
      if constexpr (OUT_F32) {
        ((float*)yv)[(size_t)row * DMODEL + col] = acc[nt][j];
      } else {
        __bf16* y = (__bf16*)yv;
        if (store_mode == 0) {
          y[(size_t)row * DMODEL + col] = (__bf16)acc[nt][j];
        } else {  // V transposed: [B, H, DK, S]
          const int bi = row / SEQ, s = row % SEQ;
          const int h = col >> 6, dk = col & 63;
          y[((size_t)(bi * NHEAD + h) * DHEAD + dk) * SEQ + s] = (__bf16)acc[nt][j];
        }
      }
    }
  }
}

// ---------------------------------------------------------------------------
// Fused attention, softmax over the HEAD axis (local per (q,k) -> no running
// stats). 16 waves = 16 heads, one 16-row q tile per workgroup.
//  - 4 K-frags loaded as one clause -> 4 score WMMAs
//  - 4 V-frags issued BEFORE the softmax LDS phase (latency hidden)
//  - cooperative head-sum: each of the 512 threads owns one (q,k) cell,
//    stores v_rcp of the 16-head sum (2KB buffer) -> frag rebuild is
//    contiguous ds_load_b128s. 34KB LDS total, no spills.
// ---------------------------------------------------------------------------
__global__ __launch_bounds__(512)
void attn_headsoftmax_kernel(const __bf16* __restrict__ Qp,
                             const __bf16* __restrict__ Kp,
                             const __bf16* __restrict__ Vt,
                             __bf16* __restrict__ Ctx) {
  __shared__ float bufE[NHEAD][16][32];  // exp(scores)        32 KB
  __shared__ float bufR[16][32];         // 1/sum over heads    2 KB

  const int lane  = threadIdx.x & 31;
  const int h     = threadIdx.x >> 5;       // wave id == head
  const int b     = blockIdx.x >> 7;        // 128 q tiles per batch
  const int qbase = (blockIdx.x & 127) * 16;
  const int m     = lane & 15;
  const int hi    = lane >> 4;
  const int sm    = threadIdx.x >> 5;       // summing thread's q row (0..15)
  const int sn    = threadIdx.x & 31;       // summing thread's k col (0..31)

  // Q A-fragments, hoisted (DK=64 -> two K=32 chunks)
  v16bf aq[2];
#pragma unroll
  for (int c = 0; c < 2; ++c) {
    const __bf16* p =
        Qp + (size_t)(b * SEQ + qbase + m) * DMODEL + h * DHEAD + c * 32;
    v8bf lo = *(const v8bf*)(p + hi * 16);
    v8bf hh = *(const v8bf*)(p + hi * 16 + 8);
#pragma unroll
    for (int i = 0; i < 8; ++i) { aq[c][i] = lo[i]; aq[c][8 + i] = hh[i]; }
  }

  v8f ctx[4] = {};
  const float c_e2 = 0.125f * 1.44269504088896f;  // (1/sqrt(DK)) * log2(e)

  for (int kb = 0; kb < SEQ; kb += 32) {
    // ---- K B-fragments: one 8x-b128 clause ----
    v16bf bk[4];  // [kt*2 + c]
#pragma unroll
    for (int kt = 0; kt < 2; ++kt)
#pragma unroll
      for (int c = 0; c < 2; ++c)
        bk[kt * 2 + c] = *(const v16bf*)(Kp +
            (size_t)(b * SEQ + kb + kt * 16 + m) * DMODEL + h * DHEAD +
            c * 32 + hi * 16);
    // ---- V B-fragments: issued now, consumed after the LDS phase ----
    v16bf bv[4];
#pragma unroll
    for (int nt = 0; nt < 4; ++nt)
      bv[nt] = *(const v16bf*)(Vt +
          ((size_t)(b * NHEAD + h) * DHEAD + nt * 16 + m) * SEQ + kb + hi * 16);

    // ---- scores: 4 WMMAs ----
    v8f s0 = {}, s1 = {};
    s0 = wmma_bf16(aq[0], bk[0], s0);
    s0 = wmma_bf16(aq[1], bk[1], s0);
    s1 = wmma_bf16(aq[0], bk[2], s1);
    s1 = wmma_bf16(aq[1], bk[3], s1);

    // ---- exp2(scale*s) -> LDS (v_mul + v_exp_f32) ----
#pragma unroll
    for (int j = 0; j < 8; ++j) {
      bufE[h][j + 8 * hi][m]      = __builtin_amdgcn_exp2f(s0[j] * c_e2);
      bufE[h][j + 8 * hi][16 + m] = __builtin_amdgcn_exp2f(s1[j] * c_e2);
    }
    if (kb + 32 < SEQ) {  // global_prefetch_b8 of next chunk
      __builtin_prefetch(Kp + (size_t)(b * SEQ + kb + 32 + m) * DMODEL + h * DHEAD, 0, 1);
      __builtin_prefetch(Vt + ((size_t)(b * NHEAD + h) * DHEAD + m) * SEQ + kb + 32, 0, 1);
    }
    __syncthreads();

    // ---- cooperative head-sum: one (q,k) cell per thread ----
    {
      float sum = 0.f;
#pragma unroll
      for (int hh = 0; hh < NHEAD; ++hh) sum += bufE[hh][sm][sn];
      bufR[sm][sn] = __builtin_amdgcn_rcpf(sum);
    }
    __syncthreads();

    // ---- rebuild attn A-fragment (contiguous b128 LDS reads) ----
    v16bf af;
#pragma unroll
    for (int i = 0; i < 8; ++i) {
      af[i]     = (__bf16)(bufE[h][m][hi * 16 + i]      * bufR[m][hi * 16 + i]);
      af[8 + i] = (__bf16)(bufE[h][m][hi * 16 + 8 + i] * bufR[m][hi * 16 + 8 + i]);
    }
    __syncthreads();  // protect bufE/bufR from next iteration's writers

    // ---- ctx += attn x V : 4 WMMAs (bv loaded long ago) ----
#pragma unroll
    for (int nt = 0; nt < 4; ++nt)
      ctx[nt] = wmma_bf16(af, bv[nt], ctx[nt]);
  }

  // ---- store ctx bf16 into [B,S,D], head-major columns ----
#pragma unroll
  for (int nt = 0; nt < 4; ++nt) {
#pragma unroll
    for (int j = 0; j < 8; ++j) {
      const int row = qbase + j + 8 * hi;
      const int col = h * DHEAD + nt * 16 + m;
      Ctx[(size_t)(b * SEQ + row) * DMODEL + col] = (__bf16)ctx[nt][j];
    }
  }
}

// ---------------------------------------------------------------------------
extern "C" void kernel_launch(void* const* d_in, const int* in_sizes, int n_in,
                              void* d_out, int out_size, void* d_ws, size_t ws_size,
                              hipStream_t stream) {
  (void)in_sizes; (void)n_in; (void)out_size; (void)ws_size;
  const float* Q  = (const float*)d_in[0];
  const float* K  = (const float*)d_in[1];
  const float* V  = (const float*)d_in[2];
  const float* Wq = (const float*)d_in[3];
  const float* Wk = (const float*)d_in[4];
  const float* Wv = (const float*)d_in[5];
  const float* Wo = (const float*)d_in[6];

  const size_t wmat = (size_t)DMODEL * DMODEL;      // 1M
  const size_t mat  = (size_t)BATCH * SEQ * DMODEL; // 4M
  __bf16* Wqb = (__bf16*)d_ws;
  __bf16* Wkb = Wqb + wmat;
  __bf16* Wvb = Wkb + wmat;
  __bf16* Wob = Wvb + wmat;
  __bf16* Qp  = Wob + wmat;
  __bf16* Kp  = Qp + mat;
  __bf16* Vt  = Kp + mat;
  __bf16* Ctx = Vt + mat;   // total 40 MB bf16 scratch

  const int packW = (int)(wmat / 8 / 256);  // 512 blocks
  pack_bf16_kernel<<<packW, 256, 0, stream>>>(Wq, Wqb, (int)wmat);
  pack_bf16_kernel<<<packW, 256, 0, stream>>>(Wk, Wkb, (int)wmat);
  pack_bf16_kernel<<<packW, 256, 0, stream>>>(Wv, Wvb, (int)wmat);
  pack_bf16_kernel<<<packW, 256, 0, stream>>>(Wo, Wob, (int)wmat);

  proj_gemm_kernel<false, false><<<512, 256, 0, stream>>>(Q, Wqb, Qp, 0);
  proj_gemm_kernel<false, false><<<512, 256, 0, stream>>>(K, Wkb, Kp, 0);
  proj_gemm_kernel<false, false><<<512, 256, 0, stream>>>(V, Wvb, Vt, 1);

  attn_headsoftmax_kernel<<<BATCH * (SEQ / 16), 512, 0, stream>>>(Qp, Kp, Vt, Ctx);

  proj_gemm_kernel<true, true><<<512, 256, 0, stream>>>(Ctx, Wob, d_out, 0);
}
